// GraphTransformerLayer_48696339202593
// MI455X (gfx1250) — matrix operand order repaired and checked by reading, not compiled, not run on hardware
//
#include <hip/hip_runtime.h>

typedef __attribute__((ext_vector_type(16))) _Float16 v16h;
typedef __attribute__((ext_vector_type(8)))  float    v8f;

#define Nn 50000
#define Ee 800000
#define Dd 128
#define Hh 8
#define DHd 16
#define DIi 64

union Frag {
    v16h h;
    unsigned int u[8];
    _Float16 e[16];
};

union H4 {
    _Float16 e[4];
    uint2 u;
};

// ---------------------------------------------------------------------------
// Reduce W_pd / W_pm over the DH axis into a padded 256x16 weight + 16 bias.
// redW[k][h] = sum_d Wsrc[k][h*16+d];   k<128 -> W_pd, k>=128 -> W_pm
// ---------------------------------------------------------------------------
__global__ __launch_bounds__(256)
void k_reduce_w(const float* __restrict__ W_pd, const float* __restrict__ b_pd,
                const float* __restrict__ W_pm, const float* __restrict__ b_pm,
                float* __restrict__ redW, float* __restrict__ redB) {
    int k = threadIdx.x; // 0..255
    const float* Wsrc = (k < Dd) ? (W_pd + (size_t)k * Dd)
                                 : (W_pm + (size_t)(k - Dd) * Dd);
    #pragma unroll
    for (int h = 0; h < 16; ++h) {
        float s = 0.0f;
        if (h < Hh) {
            #pragma unroll
            for (int d = 0; d < DHd; ++d) s += Wsrc[h * DHd + d];
        }
        redW[(size_t)k * 16 + h] = s;
    }
    if (threadIdx.x < 16) {
        int h = threadIdx.x;
        float s = 0.0f;
        if (h < Hh) {
            #pragma unroll
            for (int d = 0; d < DHd; ++d)
                s += b_pd[h * DHd + d] + b_pm[h * DHd + d];
        }
        redB[h] = s;
    }
}

// ---------------------------------------------------------------------------
// Pre-pack a row-major f32 weight W[K,Nout] into f16 WMMA B-fragments.
// Fragment f = (ct*(K/32) + kb32)*32 + lane holds 16 halfs:
//   col = ct*16 + (lane&15), k = kb32*32 + (lane&16 ? 16:0) + j
// Each lane's fragment is one contiguous 32-byte record.
// ---------------------------------------------------------------------------
__global__ __launch_bounds__(256)
void k_pack_w(const float* __restrict__ W, int K, int Nout,
              _Float16* __restrict__ Wp) {
    int tid = blockIdx.x * 256 + threadIdx.x;
    int total = (Nout >> 4) * (K >> 5) * 32;
    if (tid >= total) return;
    int lane = tid & 31;
    int rest = tid >> 5;
    int nkb  = K >> 5;
    int kb32 = rest % nkb;
    int ct   = rest / nkb;
    int col    = ct * 16 + (lane & 15);
    int kstart = kb32 * 32 + ((lane & 16) ? 16 : 0);
    _Float16* dst = Wp + (size_t)tid * 16;
    #pragma unroll
    for (int j = 0; j < 16; ++j)
        dst[j] = (_Float16)W[(size_t)(kstart + j) * Nout + col];
}

// ---------------------------------------------------------------------------
// WMMA linear: Y[M,Nout] = act( [A|A2][M,KT] @ W + bias ), W pre-packed f16.
// One block = 16 output rows staged to LDS as f16; 8 waves, one 16x16 output
// tile per wave per iteration. KT in {64,128,256} (templated -> unrolled).
// A columns k<K1 come from A (row stride K1), k>=K1 from A2 (stride KT-K1).
// ---------------------------------------------------------------------------
template <int KT>
__global__ __launch_bounds__(256)
void k_linear_wmma(const float* __restrict__ A, const float* __restrict__ A2,
                   int K1,
                   const _Float16* __restrict__ Wp, const float* __restrict__ bias,
                   float* __restrict__ Y, int Nout, int relu) {
    __shared__ _Float16 lds_a[16 * KT];

    int m0 = blockIdx.x * 16;
    int K2 = KT - K1;

    // Stage A row-block into LDS (float4 loads, f16 packed stores)
    constexpr int NV = 16 * KT / 4;
    for (int idx = threadIdx.x; idx < NV; idx += 256) {
        int r = idx / (KT / 4);
        int k = (idx - r * (KT / 4)) * 4;
        const float* srcp = (k < K1) ? (A  + (size_t)(m0 + r) * K1 + k)
                                     : (A2 + (size_t)(m0 + r) * K2 + (k - K1));
        float4 v = *(const float4*)srcp;
        H4 t;
        t.e[0] = (_Float16)v.x; t.e[1] = (_Float16)v.y;
        t.e[2] = (_Float16)v.z; t.e[3] = (_Float16)v.w;
        *(uint2*)(lds_a + r * KT + k) = t.u;
    }
    __syncthreads();

    int wave = threadIdx.x >> 5;
    int lane = threadIdx.x & 31;
    int row  = lane & 15;
    int hiHalf = (lane & 16) ? 1 : 0;
    int nt = Nout >> 4;
    constexpr int NKB = KT >> 5;

    const unsigned int* lrow = (const unsigned int*)(lds_a + row * KT);

    for (int ct = wave; ct < nt; ct += 8) {
        v8f acc = {};
        int col = ct * 16 + (lane & 15);
        const v16h* wp = (const v16h*)Wp + ((size_t)ct * NKB) * 32 + lane;
        #pragma unroll
        for (int kb32 = 0; kb32 < NKB; ++kb32) {
            // A fragment: lanes 0-15 get K=kb..kb+15,
            //             lanes 16-31 get K=kb+16..kb+31
            Frag a;
            int base = (kb32 * 32 + (hiHalf ? 16 : 0)) >> 1; // dword index
            #pragma unroll
            for (int j = 0; j < 4; ++j) {
                a.u[j]     = lrow[base + j];
                a.u[4 + j] = lrow[base + 4 + j];
            }
            // B fragment: one 32-byte contiguous load per lane
            Frag b;
            b.h = wp[kb32 * 32];
            acc = __builtin_amdgcn_wmma_f32_16x16x32_f16(
                false, a.h, false, b.h, (short)0, acc, false, false);
        }
        float bv = bias[col];
        int mhi = hiHalf ? 8 : 0;
        #pragma unroll
        for (int r8 = 0; r8 < 8; ++r8) {
            float v = acc[r8] + bv;
            if (relu) v = fmaxf(v, 0.0f);
            Y[(size_t)(m0 + mhi + r8) * Nout + col] = v;
        }
    }
}

// ---------------------------------------------------------------------------
__global__ __launch_bounds__(256)
void k_zero(float* __restrict__ p, int n) {
    int i = blockIdx.x * 256 + threadIdx.x;
    if (i < n) p[i] = 0.0f;
}

// ---------------------------------------------------------------------------
// Edge attention: one wave per edge. lane covers 4 dims (float4).
// s[h] = exp(clip(4*dot_h(K[src],Q[dst]) + eb[e,h], -5, 5))
// atomic wV[dst] += V[src]*s ; z[dst,h] += s
// ---------------------------------------------------------------------------
__global__ __launch_bounds__(256)
void k_edge(const float* __restrict__ Qb, const float* __restrict__ Kb,
            const float* __restrict__ Vb, const float* __restrict__ eb,
            const int* __restrict__ src, const int* __restrict__ dst,
            float* __restrict__ wV, float* __restrict__ z) {
    int edge = blockIdx.x * 8 + (threadIdx.x >> 5);
    if (edge >= Ee) return;
    int lane = threadIdx.x & 31;
    int s_ = src[edge], d_ = dst[edge];

    const float4 kk = ((const float4*)(Kb + (size_t)s_ * Dd))[lane];
    const float4 qq = ((const float4*)(Qb + (size_t)d_ * Dd))[lane];
    float dot = kk.x * qq.x + kk.y * qq.y + kk.z * qq.z + kk.w * qq.w;
    dot += __shfl_xor(dot, 1, 32);
    dot += __shfl_xor(dot, 2, 32);   // all 4 lanes of a head hold the head dot

    int h = lane >> 2;
    float t = 4.0f * dot + eb[(size_t)edge * 16 + h];
    t = fminf(5.0f, fmaxf(-5.0f, t));
    float sv = expf(t);

    const float4 vv = ((const float4*)(Vb + (size_t)s_ * Dd))[lane];
    float* o = wV + (size_t)d_ * Dd + lane * 4;
    atomicAdd(o + 0, vv.x * sv);
    atomicAdd(o + 1, vv.y * sv);
    atomicAdd(o + 2, vv.z * sv);
    atomicAdd(o + 3, vv.w * sv);
    if ((lane & 3) == 0) atomicAdd(z + (size_t)d_ * Hh + h, sv);
}

// ---------------------------------------------------------------------------
__global__ __launch_bounds__(256)
void k_combine(const float* __restrict__ wV, const float* __restrict__ z,
               const float* __restrict__ ib, float* __restrict__ out) {
    int i = blockIdx.x * 256 + threadIdx.x;
    if (i < Nn * Dd) {
        int n = i >> 7, d = i & 127, h = d >> 4;
        out[i] = wV[i] / (z[(size_t)n * Hh + h] + 1e-10f) + ib[i];
    }
}

// ---------------------------------------------------------------------------
// Fused residual + LayerNorm + eval-BatchNorm. One wave per 128-wide row.
// ---------------------------------------------------------------------------
__global__ __launch_bounds__(256)
void k_post_ln(const float* __restrict__ X, const float* __restrict__ R,
               const float* __restrict__ g, const float* __restrict__ b,
               const float* __restrict__ bg, const float* __restrict__ bb,
               float* __restrict__ O) {
    int row = blockIdx.x * 8 + (threadIdx.x >> 5);
    if (row >= Nn) return;
    int lane = threadIdx.x & 31;
    float4 xa = ((const float4*)(X + (size_t)row * Dd))[lane];
    float4 xb = ((const float4*)(R + (size_t)row * Dd))[lane];
    float4 x = make_float4(xa.x + xb.x, xa.y + xb.y, xa.z + xb.z, xa.w + xb.w);

    float s = x.x + x.y + x.z + x.w;
    float q = x.x * x.x + x.y * x.y + x.z * x.z + x.w * x.w;
    #pragma unroll
    for (int mm = 1; mm < 32; mm <<= 1) {
        s += __shfl_xor(s, mm, 32);
        q += __shfl_xor(q, mm, 32);
    }
    float mean = s * (1.0f / Dd);
    float var  = q * (1.0f / Dd) - mean * mean;
    float rstd = rsqrtf(var + 1e-5f);
    const float rbn = rsqrtf(1.0f + 1e-5f);

    int d0 = lane * 4;
    float4 y;
    float* yc = &y.x;
    const float* xc = &x.x;
    #pragma unroll
    for (int j = 0; j < 4; ++j) {
        int d = d0 + j;
        float v = (xc[j] - mean) * rstd * g[d] + b[d];
        yc[j] = v * rbn * bg[d] + bb[d];
    }
    ((float4*)(O + (size_t)row * Dd))[lane] = y;
}

// ---------------------------------------------------------------------------
extern "C" void kernel_launch(void* const* d_in, const int* in_sizes, int n_in,
                              void* d_out, int out_size, void* d_ws, size_t ws_size,
                              hipStream_t stream) {
    const float* h    = (const float*)d_in[0];
    const float* de   = (const float*)d_in[1];
    const float* m    = (const float*)d_in[2];
    const float* I    = (const float*)d_in[3];
    const float* W_Q  = (const float*)d_in[4];  const float* b_Q  = (const float*)d_in[5];
    const float* W_K  = (const float*)d_in[6];  const float* b_K  = (const float*)d_in[7];
    const float* W_V  = (const float*)d_in[8];  const float* b_V  = (const float*)d_in[9];
    const float* W_pd = (const float*)d_in[10]; const float* b_pd = (const float*)d_in[11];
    const float* W_pm = (const float*)d_in[12]; const float* b_pm = (const float*)d_in[13];
    const float* W_i  = (const float*)d_in[14]; const float* b_i  = (const float*)d_in[15];
    const float* W_O  = (const float*)d_in[16]; const float* b_O  = (const float*)d_in[17];
    const float* ln1g = (const float*)d_in[18]; const float* ln1b = (const float*)d_in[19];
    const float* bn1g = (const float*)d_in[20]; const float* bn1b = (const float*)d_in[21];
    const float* W_f1 = (const float*)d_in[22]; const float* b_f1 = (const float*)d_in[23];
    const float* W_f2 = (const float*)d_in[24]; const float* b_f2 = (const float*)d_in[25];
    const float* ln2g = (const float*)d_in[26]; const float* ln2b = (const float*)d_in[27];
    const float* bn2g = (const float*)d_in[28]; const float* bn2b = (const float*)d_in[29];
    const int*   src  = (const int*)d_in[30];
    const int*   dst  = (const int*)d_in[31];
    float* out = (float*)d_out;
    float* ws  = (float*)d_ws;

    const size_t ND = (size_t)Nn * Dd;       // 6,400,000
    size_t off_redW = 0;                     // 4096
    size_t off_redB = 4096;                  // 16
    size_t off_eb   = 8192;                  // E*16 = 12,800,000 (later reused as fbuf N*256)
    size_t off_Q    = off_eb + (size_t)Ee * 16;
    size_t off_K    = off_Q + ND;
    size_t off_V    = off_K + ND;
    size_t off_wV   = off_V + ND;
    size_t off_z    = off_wV + ND;           // N*8 (contiguous after wV: single zero pass)
    size_t off_bufA = off_z + (size_t)Nn * Hh;
    size_t off_bufB = off_bufA + ND;
    size_t off_pk   = off_bufB + ND;         // packed f16 weights (sizes in floats)

    float* redW = ws + off_redW;  float* redB = ws + off_redB;
    float* eb   = ws + off_eb;
    float* Qb   = ws + off_Q;     float* Kb = ws + off_K;   float* Vb = ws + off_V;
    float* wV   = ws + off_wV;    float* z  = ws + off_z;
    float* bufA = ws + off_bufA;  // Ibuf, later gbuf
    float* bufB = ws + off_bufB;  // tbuf
    float* bufC = Qb;             // obuf (Q dead after edge kernel)
    float* bufD = Kb;             // h2   (K dead after edge kernel)
    float* fbuf = eb;             // FFN hidden (eb dead after edge kernel)

    // packed weight arenas (halfs; offsets kept 64B aligned)
    _Float16* p_red = (_Float16*)(ws + off_pk);                 //  4096 halfs
    _Float16* p_WQ  = p_red + 4096;                             // 16384
    _Float16* p_WK  = p_WQ + 16384;
    _Float16* p_WV  = p_WK + 16384;
    _Float16* p_Wi  = p_WV + 16384;                             //  8192
    _Float16* p_WO  = p_Wi + 8192;                              // 16384
    _Float16* p_Wf1 = p_WO + 16384;                             // 32768
    _Float16* p_Wf2 = p_Wf1 + 32768;                            // 32768

    // 0. pack weights to f16 fragment layout (once per launch; tiny)
    k_reduce_w<<<1, 256, 0, stream>>>(W_pd, b_pd, W_pm, b_pm, redW, redB);
    k_pack_w<<<1, 256, 0, stream>>>(redW, 256, 16, p_red);
    k_pack_w<<<4, 256, 0, stream>>>(W_Q, 128, 128, p_WQ);
    k_pack_w<<<4, 256, 0, stream>>>(W_K, 128, 128, p_WK);
    k_pack_w<<<4, 256, 0, stream>>>(W_V, 128, 128, p_WV);
    k_pack_w<<<2, 256, 0, stream>>>(W_i, 64, 128, p_Wi);
    k_pack_w<<<4, 256, 0, stream>>>(W_O, 128, 128, p_WO);
    k_pack_w<<<8, 256, 0, stream>>>(W_f1, 128, 256, p_Wf1);
    k_pack_w<<<8, 256, 0, stream>>>(W_f2, 256, 128, p_Wf2);

    // 2. edge bias: eb[E,16] = [de|m] @ redW + redB   (dominant HBM stream)
    k_linear_wmma<256><<<Ee / 16, 256, 0, stream>>>(de, m, Dd, p_red, redB, eb, 16, 0);
    // 3. Q, K, V
    k_linear_wmma<128><<<Nn / 16, 256, 0, stream>>>(h, h, Dd, p_WQ, b_Q, Qb, Dd, 0);
    k_linear_wmma<128><<<Nn / 16, 256, 0, stream>>>(h, h, Dd, p_WK, b_K, Kb, Dd, 0);
    k_linear_wmma<128><<<Nn / 16, 256, 0, stream>>>(h, h, Dd, p_WV, b_V, Vb, Dd, 0);
    // 4. zero accumulators (wV + z contiguous)
    int ztot = Nn * (Dd + Hh);
    k_zero<<<(ztot + 255) / 256, 256, 0, stream>>>(wV, ztot);
    // 5. edge attention scatter (atomics land in L2: 27 MB accumulators)
    k_edge<<<Ee / 8, 256, 0, stream>>>(Qb, Kb, Vb, eb, src, dst, wV, z);
    // 6. I @ W_i
    k_linear_wmma<64><<<Nn / 16, 256, 0, stream>>>(I, I, DIi, p_Wi, b_i, bufA, Dd, 0);
    // 7. t = wV/(z+eps) + Ibuf
    k_combine<<<(Nn * Dd + 255) / 256, 256, 0, stream>>>(wV, z, bufA, bufB);
    // 8. @ W_O
    k_linear_wmma<128><<<Nn / 16, 256, 0, stream>>>(bufB, bufB, Dd, p_WO, b_O, bufC, Dd, 0);
    // 9. h2 = BN1(LN1(h + x))
    k_post_ln<<<Nn / 8, 256, 0, stream>>>(h, bufC, ln1g, ln1b, bn1g, bn1b, bufD);
    // 10. FFN up (relu)
    k_linear_wmma<128><<<Nn / 16, 256, 0, stream>>>(bufD, bufD, Dd, p_Wf1, b_f1, fbuf, 2 * Dd, 1);
    // 11. FFN down
    k_linear_wmma<256><<<Nn / 16, 256, 0, stream>>>(fbuf, fbuf, 2 * Dd, p_Wf2, b_f2, bufA, Dd, 0);
    // 12. out = BN2(LN2(h2 + x))
    k_post_ln<<<Nn / 8, 256, 0, stream>>>(bufD, bufA, ln2g, ln2b, bn2g, bn2b, out);
}
